// FullDySATModel_86260123174624
// MI455X (gfx1250) — matrix operand order repaired and checked by reading, not compiled, not run on hardware
//
#include <hip/hip_runtime.h>
#include <cstdint>

// DySAT forward for MI455X (gfx1250, wave32).
// GEMMs via v_wmma_f32_16x16x32_bf16 (f32 accum). Large bf16-A GEMMs stage
// their A panel through the Tensor Data Mover (tensor_load_to_lds -> LDS ->
// ds_load -> WMMA), tracked on TENSORcnt. Edge phase via hardware f32 atomics
// (memory/atomic bound @ 23.3 TB/s) with global_prefetch on gathered rows.

#define DEV __device__ __forceinline__

typedef __bf16 bf16_t;
typedef __attribute__((ext_vector_type(16))) __bf16 v16bf;
typedef __attribute__((ext_vector_type(8)))  float  v8f;
typedef __attribute__((ext_vector_type(4)))  unsigned u32x4;
typedef __attribute__((ext_vector_type(8)))  unsigned u32x8;

namespace {
constexpr int kN = 50000, kIN = 64, kH = 128, kSH = 8, kHD = 16;
constexpr int kTH = 4, kTHD = 32, kT = 8, kE = 400000, kC = 10;
}

DEV float elu1(float v) { return v > 0.f ? v : __expf(v) - 1.f; }

DEV float wave_sum32(float v) {
#pragma unroll
  for (int off = 16; off; off >>= 1) v += __shfl_xor(v, off, 32);
  return v;
}

DEV void atomicMaxF(float* addr, float v) {
  if (v >= 0.f) atomicMax((int*)addr, __float_as_int(v));
  else          atomicMin((unsigned int*)addr, __float_as_uint(v));
}

DEV float pos_enc(int t, int f) {
  const int i2 = f & ~1;  // 2*i
  const float freq = __expf(-(float)i2 * (0.0078125f * logf(10000.f)));  // /128
  const float ang = (float)t * freq;
  return (f & 1) ? __cosf(ang) : __sinf(ang);
}

// ---------------------------------------------------------------------------
// WMMA GEMM (fp32 A): Out[M, cols] = act(A[M,K] @ W[wrows,K]^T + bias).
// A layout (16-bit, 16x32): lane l (row = l&15, half = l>>4) holds
//   elems 0..7  -> k = half*8 + i, elems 8..15 -> k = 16 + half*8 + (i-8).
// B layout (32x16 = W^T tile): lane l (col = l&15) holds k in the same order as A.
// C/D layout: VGPR v, lane l -> m = (l>>4)*8 + v, n = l&15.
// ---------------------------------------------------------------------------
template <bool A_BF16, bool OUT_BF16, int ACT>
__global__ void k_wmma_gemm(const void* __restrict__ A_, const float* __restrict__ W,
                            const float* __restrict__ bias, void* __restrict__ Out_,
                            int M, int K, int ldo, int wrows) {
  const int wave = threadIdx.x >> 5;
  const int lane = threadIdx.x & 31;
  const int half = lane >> 4;
  const int r16  = lane & 15;
  const int row0 = (blockIdx.x * 4 + wave) * 16;
  if (row0 >= M) return;                       // M is a multiple of 16 here
  const int n0   = blockIdx.y * 16;
  const int arow = row0 + r16;
  const int wrow = n0 + r16;

  v8f acc = {};
  for (int k0 = 0; k0 < K; k0 += 32) {
    v16bf a, b;
    if (A_BF16) {
      const bf16_t* Ap = (const bf16_t*)A_ + (size_t)arow * K + k0;
#pragma unroll
      for (int i = 0; i < 8; ++i) {
        a[i]     = Ap[half * 8 + i];
        a[8 + i] = Ap[16 + half * 8 + i];
      }
    } else {
      const float* Ap = (const float*)A_ + (size_t)arow * K + k0;
#pragma unroll
      for (int i = 0; i < 8; ++i) {
        a[i]     = (bf16_t)Ap[half * 8 + i];
        a[8 + i] = (bf16_t)Ap[16 + half * 8 + i];
      }
    }
    if (wrow < wrows) {
      const float* Wr = W + (size_t)wrow * K + k0;
#pragma unroll
      for (int i = 0; i < 8; ++i) {
        b[i]     = (bf16_t)Wr[half * 8 + i];
        b[8 + i] = (bf16_t)Wr[16 + half * 8 + i];
      }
    } else {
#pragma unroll
      for (int i = 0; i < 16; ++i) b[i] = (bf16_t)0.0f;
    }
    acc = __builtin_amdgcn_wmma_f32_16x16x32_bf16(false, a, false, b, (short)0, acc,
                                                  false, false);
  }

  const int ncol = n0 + r16;
  if (ncol < wrows) {
    const float bv = (bias != nullptr) ? bias[ncol] : 0.f;
#pragma unroll
    for (int v = 0; v < 8; ++v) {
      float val = acc[v] + bv;
      if (ACT == 1) val = elu1(val);
      const size_t o = (size_t)(row0 + half * 8 + v) * (size_t)ldo + ncol;
      if (OUT_BF16) ((bf16_t*)Out_)[o] = (bf16_t)val;
      else          ((float*)Out_)[o]  = val;
    }
  }
}

// ---------------------------------------------------------------------------
// WMMA GEMM with TDM-staged A panel (bf16 A in memory, K = 128).
// One tensor_load_to_lds per block pulls the contiguous 64x128-bf16 A panel
// (16 KB) into LDS; waves then feed WMMA from LDS (ds_load_b128 runs).
// D# is a 1-D descriptor: data_size=2B, tile_dim0=8192 elements,
// tensor_dim0 = remaining elements (OOB rows of the last partial block read 0).
// ---------------------------------------------------------------------------
template <bool OUT_BF16, int ACT>
__global__ void k_wmma_gemm_tdm(const bf16_t* __restrict__ A, const float* __restrict__ W,
                                const float* __restrict__ bias, void* __restrict__ Out_,
                                int M, int ldo, int wrows, unsigned tensor_elems) {
  constexpr int K = kH;  // 128
  __shared__ bf16_t lds_a[64 * K];
  const int row0blk = blockIdx.x * 64;

  if (threadIdx.x < 32) {  // wave 0 issues the TDM DMA (EXEC ignored by TDM)
    const unsigned long long gaddr =
        (unsigned long long)(const void*)(A + (size_t)row0blk * K);
    const unsigned ldsaddr = (unsigned)(uintptr_t)&lds_a[0];
    const unsigned nelem   = 64u * (unsigned)K;  // 8192 bf16 = 16 KB tile
    const unsigned remain  = tensor_elems - (unsigned)row0blk * (unsigned)K;
    u32x4 g0;
    g0[0] = 1u;                                                   // count=1, user D#
    g0[1] = ldsaddr;                                              // lds_addr
    g0[2] = (unsigned)gaddr;                                      // global_addr[31:0]
    g0[3] = (unsigned)((gaddr >> 32) & 0x01FFFFFFu) | (2u << 30); // [56:32] | type=2
    u32x8 g1;
    g1[0] = 0x00010000u;                 // data_size=1 (2 bytes), no multicast
    g1[1] = (remain & 0xFFFFu) << 16;    // tensor_dim0[15:0]
    g1[2] = (remain >> 16) | (1u << 16); // tensor_dim0[31:16], tensor_dim1=1
    g1[3] = (nelem & 0xFFFFu) << 16;     // tile_dim0 = 8192
    g1[4] = 0u;                          // tile_dim1/tile_dim2 unused
    g1[5] = remain;                      // tensor_dim0_stride[31:0]
    g1[6] = 0u;
    g1[7] = 0u;
    asm volatile("tensor_load_to_lds %0, %1" ::"s"(g0), "s"(g1) : "memory");
    __builtin_amdgcn_s_wait_tensorcnt(0);
  }
  __syncthreads();

  const int wave = threadIdx.x >> 5;
  const int lane = threadIdx.x & 31;
  const int half = lane >> 4;
  const int r16  = lane & 15;
  const int n0   = blockIdx.y * 16;
  const int wrow = n0 + r16;
  const bf16_t* Arow = lds_a + (size_t)(wave * 16 + r16) * K;

  v8f acc = {};
  for (int k0 = 0; k0 < K; k0 += 32) {
    v16bf a, b;
#pragma unroll
    for (int i = 0; i < 8; ++i) {
      a[i]     = Arow[k0 + half * 8 + i];
      a[8 + i] = Arow[k0 + 16 + half * 8 + i];
    }
    if (wrow < wrows) {
      const float* Wr = W + (size_t)wrow * K + k0;
#pragma unroll
      for (int i = 0; i < 8; ++i) {
        b[i]     = (bf16_t)Wr[half * 8 + i];
        b[8 + i] = (bf16_t)Wr[16 + half * 8 + i];
      }
    } else {
#pragma unroll
      for (int i = 0; i < 16; ++i) b[i] = (bf16_t)0.0f;
    }
    acc = __builtin_amdgcn_wmma_f32_16x16x32_bf16(false, a, false, b, (short)0, acc,
                                                  false, false);
  }

  const int row0 = row0blk + wave * 16;
  const int ncol = n0 + r16;
  if (ncol < wrows) {
    const float bv = (bias != nullptr) ? bias[ncol] : 0.f;
#pragma unroll
    for (int v = 0; v < 8; ++v) {
      const int orow = row0 + half * 8 + v;
      if (orow < M) {
        float val = acc[v] + bv;
        if (ACT == 1) val = elu1(val);
        const size_t o = (size_t)orow * (size_t)ldo + ncol;
        if (OUT_BF16) ((bf16_t*)Out_)[o] = (bf16_t)val;
        else          ((float*)Out_)[o]  = val;
      }
    }
  }
}

// --------------------------- GAT edge phase --------------------------------
// s[e,h] = leakyrelu(p1[src,h] + p2[dst,h]) with
//   p1[n,h] = h[n,h,:]·a[:16],  p2[n,h] = h[n,h,:]·a[16:]   (snapshot-invariant)
__global__ void k_precompute_p(const float* __restrict__ h, const float* __restrict__ a,
                               float* __restrict__ p1, float* __restrict__ p2, int n) {
  int idx = blockIdx.x * blockDim.x + threadIdx.x;
  if (idx >= n * kSH) return;
  const int node = idx >> 3, hh = idx & 7;
  const float* hp = h + (size_t)node * kH + hh * kHD;
  float s1 = 0.f, s2 = 0.f;
#pragma unroll
  for (int d = 0; d < kHD; ++d) {
    s1 += hp[d] * a[d];
    s2 += hp[d] * a[kHD + d];
  }
  p1[idx] = s1;
  p2[idx] = s2;
}

__global__ void k_gat_init(float* __restrict__ m, float* __restrict__ z,
                           float* __restrict__ gacc, int n) {
  int idx = blockIdx.x * blockDim.x + threadIdx.x;
  if (idx < n * kSH) { m[idx] = -3.0e38f; z[idx] = 0.f; }
  if (idx < n * kH) gacc[idx] = 0.f;
}

__global__ void k_edge_score(const int* __restrict__ ei, const float* __restrict__ p1,
                             const float* __restrict__ p2, float* __restrict__ sbuf,
                             float* __restrict__ m, int e_cnt) {
  int idx = blockIdx.x * blockDim.x + threadIdx.x;
  if (idx >= e_cnt * kSH) return;
  const int e = idx >> 3, hh = idx & 7;
  const int src = ei[e], dst = ei[e_cnt + e];
  float s = p1[src * kSH + hh] + p2[dst * kSH + hh];
  s = s > 0.f ? s : 0.2f * s;  // LeakyReLU(0.2)
  sbuf[idx] = s;
  atomicMaxF(&m[dst * kSH + hh], s);
}

__global__ void k_edge_expsum(const int* __restrict__ ei, float* __restrict__ sbuf,
                              const float* __restrict__ m, float* __restrict__ z,
                              int e_cnt) {
  int idx = blockIdx.x * blockDim.x + threadIdx.x;
  if (idx >= e_cnt * kSH) return;
  const int e = idx >> 3, hh = idx & 7;
  const int dst = ei[e_cnt + e];
  const float ev = __expf(sbuf[idx] - m[dst * kSH + hh]);
  sbuf[idx] = ev;
  unsafeAtomicAdd(&z[dst * kSH + hh], ev);  // global_atomic_add_f32
}

__global__ void k_edge_message(const int* __restrict__ ei, const float* __restrict__ sbuf,
                               const float* __restrict__ z, const float* __restrict__ h,
                               float* __restrict__ gacc, int e_cnt) {
  int idx = blockIdx.x * blockDim.x + threadIdx.x;
  if (idx >= e_cnt * kSH) return;
  const int e = idx >> 3, hh = idx & 7;
  const int src = ei[e], dst = ei[e_cnt + e];
  const float* hrow = h + (size_t)src * kH + hh * kHD;
  __builtin_prefetch(hrow, 0, 0);  // global_prefetch: hide the random gather
  const float alpha = sbuf[idx] / z[dst * kSH + hh];
  const float4* hs = (const float4*)hrow;
  float* gp = gacc + (size_t)dst * kH + hh * kHD;
#pragma unroll
  for (int q4 = 0; q4 < 4; ++q4) {
    const float4 hv = hs[q4];
    unsafeAtomicAdd(gp + q4 * 4 + 0, hv.x * alpha);
    unsafeAtomicAdd(gp + q4 * 4 + 1, hv.y * alpha);
    unsafeAtomicAdd(gp + q4 * 4 + 2, hv.z * alpha);
    unsafeAtomicAdd(gp + q4 * 4 + 3, hv.w * alpha);
  }
}

// seqpe[t,n,:] = bf16( elu(LN1(gacc + x)) + PE[t,:] ); one wave32 per node row.
__global__ void k_gat_ln(const float* __restrict__ gacc, const float* __restrict__ x,
                         const float* __restrict__ g, const float* __restrict__ b,
                         bf16_t* __restrict__ seqpe_t, int t, int n) {
  const int node = blockIdx.x * 4 + (threadIdx.x >> 5);
  if (node >= n) return;
  const int lane = threadIdx.x & 31;
  const float4 gv = ((const float4*)(gacc + (size_t)node * kH))[lane];
  const float4 xv = ((const float4*)(x + (size_t)node * kH))[lane];
  float v[4] = {gv.x + xv.x, gv.y + xv.y, gv.z + xv.z, gv.w + xv.w};
  const float mu = wave_sum32(v[0] + v[1] + v[2] + v[3]) * (1.f / 128.f);
  float d0 = v[0] - mu, d1 = v[1] - mu, d2 = v[2] - mu, d3 = v[3] - mu;
  const float var = wave_sum32(d0 * d0 + d1 * d1 + d2 * d2 + d3 * d3) * (1.f / 128.f);
  const float rs = rsqrtf(var + 1e-5f);
#pragma unroll
  for (int c = 0; c < 4; ++c) {
    const int f = lane * 4 + c;
    float y = (v[c] - mu) * rs * g[f] + b[f];
    y = elu1(y) + pos_enc(t, f);
    seqpe_t[(size_t)node * kH + f] = (bf16_t)y;
  }
}

// Temporal attention at t = T-1 only (classifier uses out[-1]).
// Block = 128 threads = 4 waves = 4 heads; lane = head-dim element.
__global__ void k_temporal_attn(const float* __restrict__ q, const bf16_t* __restrict__ kv,
                                float* __restrict__ o, int n) {
  const int node = blockIdx.x;
  if (node >= n) return;
  const int head = threadIdx.x >> 5;
  const int lane = threadIdx.x & 31;
  const float qd = q[(size_t)node * kH + head * kTHD + lane] * 0.17677669529663687f; // 1/sqrt(32)
  float sc[kT];
#pragma unroll
  for (int t = 0; t < kT; ++t) {
    const float kd = (float)kv[((size_t)t * kN + node) * (2 * kH) + head * kTHD + lane];
    sc[t] = wave_sum32(qd * kd);  // broadcast full dot to all lanes
  }
  float mx = sc[0];
#pragma unroll
  for (int t = 1; t < kT; ++t) mx = fmaxf(mx, sc[t]);
  float den = 0.f;
#pragma unroll
  for (int t = 0; t < kT; ++t) { sc[t] = __expf(sc[t] - mx); den += sc[t]; }
  const float inv = 1.f / den;
  float od = 0.f;
#pragma unroll
  for (int t = 0; t < kT; ++t) {
    const float vd = (float)kv[((size_t)t * kN + node) * (2 * kH) + kH + head * kTHD + lane];
    od += sc[t] * inv * vd;
  }
  o[(size_t)node * kH + head * kTHD + lane] = od;
}

// final[n,:] = elu(LN2(seqpe[T-1] + oproj))
__global__ void k_final_ln(const bf16_t* __restrict__ seq7, const float* __restrict__ op,
                           const float* __restrict__ g, const float* __restrict__ b,
                           float* __restrict__ out, int n) {
  const int node = blockIdx.x * 4 + (threadIdx.x >> 5);
  if (node >= n) return;
  const int lane = threadIdx.x & 31;
  const float4 ov = ((const float4*)(op + (size_t)node * kH))[lane];
  const bf16_t* sp = seq7 + (size_t)node * kH + lane * 4;
  float v[4] = {(float)sp[0] + ov.x, (float)sp[1] + ov.y, (float)sp[2] + ov.z,
                (float)sp[3] + ov.w};
  const float mu = wave_sum32(v[0] + v[1] + v[2] + v[3]) * (1.f / 128.f);
  float d0 = v[0] - mu, d1 = v[1] - mu, d2 = v[2] - mu, d3 = v[3] - mu;
  const float var = wave_sum32(d0 * d0 + d1 * d1 + d2 * d2 + d3 * d3) * (1.f / 128.f);
  const float rs = rsqrtf(var + 1e-5f);
#pragma unroll
  for (int c = 0; c < 4; ++c) {
    const int f = lane * 4 + c;
    out[(size_t)node * kH + f] = elu1((v[c] - mu) * rs * g[f] + b[f]);
  }
}

// ---------------------------------------------------------------------------
extern "C" void kernel_launch(void* const* d_in, const int* in_sizes, int n_in,
                              void* d_out, int out_size, void* d_ws, size_t ws_size,
                              hipStream_t stream) {
  (void)in_sizes; (void)n_in; (void)out_size; (void)ws_size;
  const float* x_orig = (const float*)d_in[0];
  const int*   ei     = (const int*)d_in[1];   // [T,2,E]
  // d_in[2] node_masks: all-ones in eval -> identity, unused
  const float* Wp   = (const float*)d_in[3];
  const float* bp   = (const float*)d_in[4];
  const float* Wg   = (const float*)d_in[5];
  const float* aat  = (const float*)d_in[6];
  const float* ln1g = (const float*)d_in[7];
  const float* ln1b = (const float*)d_in[8];
  const float* Wqkv = (const float*)d_in[9];
  const float* bqkv = (const float*)d_in[10];
  const float* Wo   = (const float*)d_in[11];
  const float* bo   = (const float*)d_in[12];
  const float* ln2g = (const float*)d_in[13];
  const float* ln2b = (const float*)d_in[14];
  const float* Wc   = (const float*)d_in[15];
  const float* bc   = (const float*)d_in[16];
  float* logits = (float*)d_out;

  char* wsp = (char*)d_ws;
  size_t off = 0;
  auto take = [&](size_t bytes) {
    void* p = wsp + off;
    off = (off + bytes + 255) & ~(size_t)255;
    return p;
  };
  const size_t NH4 = (size_t)kN * kH * 4;
  float*  x     = (float*)take(NH4);
  float*  hbuf  = (float*)take(NH4);
  float*  p1    = (float*)take((size_t)kN * kSH * 4);
  float*  p2    = (float*)take((size_t)kN * kSH * 4);
  float*  mbuf  = (float*)take((size_t)kN * kSH * 4);
  float*  zbuf  = (float*)take((size_t)kN * kSH * 4);
  float*  sbuf  = (float*)take((size_t)kE * kSH * 4);
  float*  gacc  = (float*)take(NH4);
  bf16_t* seqpe = (bf16_t*)take((size_t)kT * kN * kH * 2);
  bf16_t* kvbuf = (bf16_t*)take((size_t)kT * kN * 2 * kH * 2);
  float*  qbuf  = (float*)take(NH4);
  float*  obuf  = (float*)take(NH4);
  float*  oproj = (float*)take(NH4);
  float*  fin   = (float*)take(NH4);

  const dim3 blk(128);

  // 1) x = elu(x_orig @ Wp^T + bp)            [N,64]x[128,64]^T
  k_wmma_gemm<false, false, 1><<<dim3((kN + 63) / 64, kH / 16), blk, 0, stream>>>(
      x_orig, Wp, bp, x, kN, kIN, kH, kH);

  // 2) h = x @ Wg^T (snapshot-invariant — computed once, not per snapshot)
  k_wmma_gemm<false, false, 0><<<dim3((kN + 63) / 64, kH / 16), blk, 0, stream>>>(
      x, Wg, nullptr, hbuf, kN, kH, kH, kH);

  // 3) per-node attention score halves (snapshot-invariant)
  k_precompute_p<<<(kN * kSH + 255) / 256, 256, 0, stream>>>(hbuf, aat, p1, p2, kN);

  // 4) per-snapshot GAT: scatter softmax + scatter-add messages + LN/ELU/PE
  for (int t = 0; t < kT; ++t) {
    const int* eit = ei + (size_t)t * 2 * kE;
    k_gat_init<<<(kN * kH + 255) / 256, 256, 0, stream>>>(mbuf, zbuf, gacc, kN);
    k_edge_score<<<(kE * kSH + 255) / 256, 256, 0, stream>>>(eit, p1, p2, sbuf, mbuf, kE);
    k_edge_expsum<<<(kE * kSH + 255) / 256, 256, 0, stream>>>(eit, sbuf, mbuf, zbuf, kE);
    k_edge_message<<<(kE * kSH + 255) / 256, 256, 0, stream>>>(eit, sbuf, zbuf, hbuf, gacc, kE);
    k_gat_ln<<<(kN + 3) / 4, blk, 0, stream>>>(gacc, x, ln1g, ln1b,
                                               seqpe + (size_t)t * kN * kH, t, kN);
  }

  // 5) k,v for all T via TDM-staged WMMA: kv = seqpe @ Wqkv[128:384]^T + bqkv[128:]
  k_wmma_gemm_tdm<true, 0><<<dim3((kT * kN) / 64, (2 * kH) / 16), blk, 0, stream>>>(
      seqpe, Wqkv + (size_t)kH * kH, bqkv + kH, kvbuf, kT * kN, 2 * kH, 2 * kH,
      (unsigned)(kT * kN) * (unsigned)kH);

  // 6) q only at t = T-1 (only out[-1] reaches the classifier), TDM-staged
  k_wmma_gemm_tdm<false, 0><<<dim3((kN + 63) / 64, kH / 16), blk, 0, stream>>>(
      seqpe + (size_t)(kT - 1) * kN * kH, Wqkv, bqkv, qbuf, kN, kH, kH,
      (unsigned)kN * (unsigned)kH);

  // 7) temporal attention over T=8 per (node, head)
  k_temporal_attn<<<kN, blk, 0, stream>>>(qbuf, kvbuf, obuf, kN);

  // 8) oproj = o @ Wo^T + bo
  k_wmma_gemm<false, false, 0><<<dim3((kN + 63) / 64, kH / 16), blk, 0, stream>>>(
      obuf, Wo, bo, oproj, kN, kH, kH, kH);

  // 9) final = elu(LN2(seqpe[T-1] + oproj))
  k_final_ln<<<(kN + 3) / 4, blk, 0, stream>>>(
      seqpe + (size_t)(kT - 1) * kN * kH, oproj, ln2g, ln2b, fin, kN);

  // 10) logits = final @ Wc^T + bc   (wrows=10 pads the 16-wide WMMA tile)
  k_wmma_gemm<false, false, 0><<<dim3((kN + 63) / 64, 1), blk, 0, stream>>>(
      fin, Wc, bc, logits, kN, kH, kC, kC);
}
